// PAM_Module_5025111736716
// MI455X (gfx1250) — matrix operand order repaired and checked by reading, not compiled, not run on hardware
//
#include <hip/hip_runtime.h>
#include <hip/hip_bf16.h>
#include <stdint.h>

// Problem dims
#define Bn 8
#define Cc 512
#define CId 64
#define Nn 4096

typedef __bf16 bf16;
typedef __attribute__((ext_vector_type(16))) __bf16 bf16x16;
typedef __attribute__((ext_vector_type(8)))  __bf16 bf16x8;
typedef __attribute__((ext_vector_type(8)))  float  f32x8;

__device__ __forceinline__ f32x8 zero8() {
  f32x8 z;
#pragma unroll
  for (int i = 0; i < 8; ++i) z[i] = 0.0f;
  return z;
}

__device__ __forceinline__ f32x8 wmma_bf16(bf16x16 a, bf16x16 b, f32x8 c) {
  // D = A(16x32) * B(32x16) + C, bf16 in / f32 accumulate
  return __builtin_amdgcn_wmma_f32_16x16x32_bf16(false, a, false, b, (short)0, c,
                                                 false, false);
}

// A-matrix tile (16 rows x 32 K), source row-major with leading dim `ld` (elems).
// Lane m (0..15) holds row m; lane group kg=lane>>4 selects K halves:
//   elems 0..7  -> K = 8*kg + 0..7 ; elems 8..15 -> K = 16 + 8*kg + 0..7
__device__ __forceinline__ bf16x16 load_tile_A(const bf16* src, int ld, int lane) {
  const int r = lane & 15, kg = lane >> 4;
  const bf16* p = src + (size_t)r * ld + 8 * kg;
  bf16x8 lo = *(const bf16x8*)p;
  bf16x8 hi = *(const bf16x8*)(p + 16);
  bf16x16 v;
#pragma unroll
  for (int e = 0; e < 8; ++e) { v[e] = lo[e]; v[e + 8] = hi[e]; }
  return v;
}

// B-matrix tile (32 K x 16 cols), source stored as src[col][K] row-major, ld elems.
// Lane col=lane&15, kg=lane>>4: elems 0..7 -> K = 8*kg + 0..7 ; elems 8..15 -> K = 16 + 8*kg + 0..7
__device__ __forceinline__ bf16x16 load_tile_B(const bf16* src, int ld, int lane) {
  const int c = lane & 15, kg = lane >> 4;
  const bf16* p = src + (size_t)c * ld + 8 * kg;
  bf16x8 lo = *(const bf16x8*)p;
  bf16x8 hi = *(const bf16x8*)(p + 16);
  bf16x16 v;
#pragma unroll
  for (int e = 0; e < 8; ++e) { v[e] = lo[e]; v[e + 8] = hi[e]; }
  return v;
}

// -------- x (B,C,N) f32 -> xT (B,N,C) bf16, tiled transpose through LDS --------
__global__ void cvt_x_transpose(const float* __restrict__ x, bf16* __restrict__ xbT) {
  __shared__ float tile[32][33];
  const int b = blockIdx.z;
  const int n0 = blockIdx.x * 32, c0 = blockIdx.y * 32;
  const int tx = threadIdx.x, ty = threadIdx.y;  // 32 x 8
  const float* src = x + (size_t)b * Cc * Nn;
#pragma unroll
  for (int i = 0; i < 4; ++i)
    tile[ty + 8 * i][tx] = src[(size_t)(c0 + ty + 8 * i) * Nn + n0 + tx];
  __syncthreads();
  bf16* dst = xbT + (size_t)b * Nn * Cc;
#pragma unroll
  for (int i = 0; i < 4; ++i)
    dst[(size_t)(n0 + ty + 8 * i) * Cc + c0 + tx] = (bf16)tile[tx][ty + 8 * i];
}

__global__ void cvt_w(const float* __restrict__ s, bf16* __restrict__ d, int n) {
  int i = blockIdx.x * blockDim.x + threadIdx.x;
  if (i < n) d[i] = (bf16)s[i];
}

// -------- q/k projections: qT[b][n][d] = sum_c wq[d][c]*x[c][n] + bq[d] --------
// grid (Nn/64, Bn), 256 thr = 8 waves: waves 0..3 -> q, 4..7 -> k, nTile = w&3.
// Full operand set (4 A tiles + 1 B tile) batch double-buffered across K so all
// next-iteration loads are outstanding while the current 4 WMMAs execute.
__global__ void __launch_bounds__(256) proj_qk(
    const bf16* __restrict__ xbT, const bf16* __restrict__ wqb,
    const float* __restrict__ bq, const bf16* __restrict__ wkb,
    const float* __restrict__ bk, bf16* __restrict__ qT, bf16* __restrict__ kT) {
  const int b = blockIdx.y;
  const int n0 = blockIdx.x * 64;
  const int w = threadIdx.x >> 5, lane = threadIdx.x & 31;
  const bool is_k = (w >= 4);
  const int nt = w & 3;
  const bf16* wb = is_k ? wkb : wqb;
  const float* bias = is_k ? bk : bq;
  bf16* outp = (is_k ? kT : qT) + (size_t)b * Nn * CId;
  const bf16* xrow = xbT + ((size_t)b * Nn + n0 + 16 * nt) * Cc;

  f32x8 acc[4];
#pragma unroll
  for (int dt = 0; dt < 4; ++dt) acc[dt] = zero8();

  bf16x16 At[4], AtN[4];
  bf16x16 Bt = load_tile_B(xrow, Cc, lane);
#pragma unroll
  for (int dt = 0; dt < 4; ++dt)
    At[dt] = load_tile_A(wb + (size_t)(16 * dt) * Cc, Cc, lane);

  for (int k0 = 0; k0 < Cc; k0 += 32) {
    const int kn = (k0 + 32 < Cc) ? k0 + 32 : k0;   // next-iter operands in flight
    bf16x16 Bnx = load_tile_B(xrow + kn, Cc, lane);
#pragma unroll
    for (int dt = 0; dt < 4; ++dt)
      AtN[dt] = load_tile_A(wb + (size_t)(16 * dt) * Cc + kn, Cc, lane);
#pragma unroll
    for (int dt = 0; dt < 4; ++dt) acc[dt] = wmma_bf16(At[dt], Bt, acc[dt]);
#pragma unroll
    for (int dt = 0; dt < 4; ++dt) At[dt] = AtN[dt];
    Bt = Bnx;
  }
  const int g = lane >> 4, col = lane & 15;
  const int n = n0 + 16 * nt + col;
#pragma unroll
  for (int dt = 0; dt < 4; ++dt) {
    bf16x8 o;
#pragma unroll
    for (int r = 0; r < 8; ++r) o[r] = (bf16)(acc[dt][r] + bias[16 * dt + 8 * g + r]);
    *(bf16x8*)(outp + (size_t)n * CId + 16 * dt + 8 * g) = o;  // contiguous in d
  }
}

// -------- v projection: vD[b][c][n] = sum wv[c][:]*x[:][n] + bv[c] --------
// grid (Nn/64, Cc/128, Bn), 256 thr: wave w -> c tile (c0+16w), nt 0..3.
// Batch double-buffered operands (1 A + 4 B tiles).
__global__ void __launch_bounds__(256) proj_v(
    const bf16* __restrict__ xbT, const bf16* __restrict__ wvb,
    const float* __restrict__ bv, bf16* __restrict__ vD) {
  const int b = blockIdx.z;
  const int n0 = blockIdx.x * 64;
  const int c0 = blockIdx.y * 128;
  const int w = threadIdx.x >> 5, lane = threadIdx.x & 31;
  const int crow = c0 + 16 * w;

  f32x8 acc[4];
#pragma unroll
  for (int nt = 0; nt < 4; ++nt) acc[nt] = zero8();

  const bf16* arow = wvb + (size_t)crow * Cc;
  const bf16* xbase = xbT + (size_t)b * Nn * Cc;

  bf16x16 Bt[4], BtN[4];
  bf16x16 At = load_tile_A(arow, Cc, lane);
#pragma unroll
  for (int nt = 0; nt < 4; ++nt)
    Bt[nt] = load_tile_B(xbase + (size_t)(n0 + 16 * nt) * Cc, Cc, lane);

  for (int k0 = 0; k0 < Cc; k0 += 32) {
    const int kn = (k0 + 32 < Cc) ? k0 + 32 : k0;
    bf16x16 Anx = load_tile_A(arow + kn, Cc, lane);
#pragma unroll
    for (int nt = 0; nt < 4; ++nt)
      BtN[nt] = load_tile_B(xbase + (size_t)(n0 + 16 * nt) * Cc + kn, Cc, lane);
#pragma unroll
    for (int nt = 0; nt < 4; ++nt) acc[nt] = wmma_bf16(At, Bt[nt], acc[nt]);
#pragma unroll
    for (int nt = 0; nt < 4; ++nt) Bt[nt] = BtN[nt];
    At = Anx;
  }
  const int g = lane >> 4, col = lane & 15;
  bf16* outp = vD + (size_t)b * Cc * Nn;
#pragma unroll
  for (int nt = 0; nt < 4; ++nt) {
    const int i = n0 + 16 * nt + col;
#pragma unroll
    for (int r = 0; r < 8; ++r) {
      const int c = crow + 8 * g + r;
      outp[(size_t)c * Nn + i] = (bf16)(acc[nt][r] + bv[c]);
    }
  }
}

// -------- pass 1: per-row softmax stats over e = qT * kT^T (online max/sum) ----
// grid (Nn/128, Bn), 256 thr: wave w owns 16 rows i0 = blk*128 + 16w, streams j.
// k tiles double-buffered: next pair issued before the current WMMAs.
__global__ void __launch_bounds__(256) softmax_stats(
    const bf16* __restrict__ qT, const bf16* __restrict__ kT,
    float2* __restrict__ stats) {
  const int b = blockIdx.y;
  const int w = threadIdx.x >> 5, lane = threadIdx.x & 31;
  const int i0 = blockIdx.x * 128 + 16 * w;
  const bf16* qrow = qT + ((size_t)b * Nn + i0) * CId;
  bf16x16 A0 = load_tile_A(qrow, CId, lane);        // K = d 0..31
  bf16x16 A1 = load_tile_A(qrow + 32, CId, lane);   // K = d 32..63

  float m[8], s[8];
#pragma unroll
  for (int r = 0; r < 8; ++r) { m[r] = -3.0e38f; s[r] = 0.0f; }

  const bf16* kbase = kT + (size_t)b * Nn * CId;
  const int JT = Nn / 16;
  bf16x16 B0 = load_tile_B(kbase, CId, lane);
  bf16x16 B1 = load_tile_B(kbase + 32, CId, lane);
  for (int jt = 0; jt < JT; ++jt) {
    const int jn = (jt + 1 < JT) ? jt + 1 : jt;
    const bf16* kp = kbase + (size_t)(jn * 16) * CId;
    bf16x16 N0 = load_tile_B(kp, CId, lane);        // next tiles, stay in flight
    bf16x16 N1 = load_tile_B(kp + 32, CId, lane);
    f32x8 e = zero8();
    e = wmma_bf16(A0, B0, e);
    e = wmma_bf16(A1, B1, e);
#pragma unroll
    for (int r = 0; r < 8; ++r) {  // online softmax update, 16 j's per row live
      const float ev = e[r];
      const float mn = fmaxf(m[r], ev);
      s[r] = s[r] * __expf(m[r] - mn) + __expf(ev - mn);
      m[r] = mn;
    }
    B0 = N0; B1 = N1;
  }
  // reduce the 16 column slots of each row (lanes within each 16-group)
#pragma unroll
  for (int off = 1; off < 16; off <<= 1) {
#pragma unroll
    for (int r = 0; r < 8; ++r) {
      const float mo = __shfl_xor(m[r], off, 32);
      const float so = __shfl_xor(s[r], off, 32);
      const float mn = fmaxf(m[r], mo);
      s[r] = s[r] * __expf(m[r] - mn) + so * __expf(mo - mn);
      m[r] = mn;
    }
  }
  if ((lane & 15) == 0) {
    const int g = lane >> 4;
#pragma unroll
    for (int r = 0; r < 8; ++r) {
      const int i = i0 + 8 * g + r;
      stats[(size_t)b * Nn + i] = make_float2(m[r], 1.0f / s[r]);
    }
  }
}

// -------- pass 2: fused out[c][j] = sum_i v[c][i] * exp(e[i][j]-m_i)*rZ_i ------
// grid (Nn/64 jblk, Cc/256 cblk, Bn), 256 thr / 8 waves.
// Per i-step (32): waves compute the 32x64 attn tile via WMMA (+exp scale),
// round-trip through parity-double-buffered LDS to become B of the out GEMM.
// Each wave owns two M tiles (c0+16w, c0+128+16w) -> e recompute only 2x.
__global__ void __launch_bounds__(256) attn_out(
    const bf16* __restrict__ qT, const bf16* __restrict__ kT,
    const bf16* __restrict__ vD, const float2* __restrict__ stats,
    const float* __restrict__ x, const float* __restrict__ gamma_p,
    float* __restrict__ out) {
  __shared__ bf16 kS[64 * CId];          // k tile for this j-block (8 KB), loaded once
  __shared__ bf16 attnS[2][64 * 40];     // attn tile [j][i], ld=40, parity buffered

  const int b = blockIdx.z;
  const int j0 = blockIdx.x * 64;
  const int c0 = blockIdx.y * 256;
  const int w = threadIdx.x >> 5, lane = threadIdx.x & 31;

  // --- async-stage kT rows j0..j0+63 into LDS (CDNA5 async global->LDS) ---
  {
    const int t = threadIdx.x;                       // 256 thr * 32 B = 8 KB
    const bf16* src = kT + ((size_t)b * Nn + j0) * CId;
    unsigned lds0 = ((unsigned)(uintptr_t)(&kS[0])) + (unsigned)(t * 32);
    unsigned long long g0 =
        (unsigned long long)(uintptr_t)src + (unsigned long long)(t * 32);
    asm volatile(
        "global_load_async_to_lds_b128 %0, %1, off\n\t"
        "global_load_async_to_lds_b128 %0, %1, off offset:16"
        :: "v"(lds0), "v"(g0) : "memory");
    asm volatile("s_wait_asynccnt 0" ::: "memory");
  }
  __syncthreads();

  const int iTile = w >> 2;   // e-tile row (0..1)
  const int jTile = w & 3;    // e-tile col (0..3)
  const int g = lane >> 4, col = lane & 15;

  // loop-invariant k B-operands for this wave's e tile (registers, not re-read)
  bf16x16 Bk0 = load_tile_B(kS + (size_t)(16 * jTile) * CId, CId, lane);
  bf16x16 Bk1 = load_tile_B(kS + (size_t)(16 * jTile) * CId + 32, CId, lane);

  f32x8 oacc[2][4];
#pragma unroll
  for (int mg = 0; mg < 2; ++mg)
#pragma unroll
    for (int nt = 0; nt < 4; ++nt) oacc[mg][nt] = zero8();

  const bf16* qbase = qT + (size_t)b * Nn * CId;
  const bf16* vb0 = vD + (size_t)b * Cc * Nn + (size_t)(c0 + 16 * w) * Nn;
  const bf16* vb1 = vb0 + (size_t)128 * Nn;
  const float2* st = stats + (size_t)b * Nn;

  // double-buffered q A-operands across i-steps
  bf16x16 Aq0 = load_tile_A(qbase + (size_t)(16 * iTile) * CId, CId, lane);
  bf16x16 Aq1 = load_tile_A(qbase + (size_t)(16 * iTile) * CId + 32, CId, lane);

  for (int i0 = 0; i0 < Nn; i0 += 32) {
    // hoist this step's v A-operands early so they cover the e-phase latency
    bf16x16 Av0 = load_tile_A(vb0 + i0, Nn, lane);
    bf16x16 Av1 = load_tile_A(vb1 + i0, Nn, lane);

    // ---- e tile (this wave: rows i0+16*iTile.., cols j0+16*jTile..) ----
    f32x8 e = zero8();
    e = wmma_bf16(Aq0, Bk0, e);
    e = wmma_bf16(Aq1, Bk1, e);

    // prefetch next i-step's q A-operands (stay in flight through out-phase)
    const int in_ = (i0 + 32 < Nn) ? i0 + 32 : i0;
    const bf16* qrn = qbase + (size_t)(in_ + 16 * iTile) * CId;
    bf16x16 AqN0 = load_tile_A(qrn, CId, lane);
    bf16x16 AqN1 = load_tile_A(qrn + 32, CId, lane);

    // ---- attn = exp(e - m_i) * rZ_i, packed to bf16, stored [j][i] in LDS ----
    const int par = (i0 >> 5) & 1;
    bf16* aw = attnS[par];
    bf16x8 av;
#pragma unroll
    for (int r = 0; r < 8; ++r) {
      const int irow = i0 + 16 * iTile + 8 * g + r;
      const float2 ms = st[irow];
      av[r] = (bf16)(__expf(e[r] - ms.x) * ms.y);
    }
    *(bf16x8*)(aw + (size_t)(16 * jTile + col) * 40 + 16 * iTile + 8 * g) = av;
    __syncthreads();   // single barrier per i-step (parity buffer handles WAR)

    // ---- out GEMM step: A = v rows (2 M tiles), B = attnS[par] ----
    bf16x16 Ba[4];     // batch the LDS reads ahead of the 8 WMMAs
#pragma unroll
    for (int nt = 0; nt < 4; ++nt)
      Ba[nt] = load_tile_B(aw + (size_t)(16 * nt) * 40, 40, lane);
#pragma unroll
    for (int nt = 0; nt < 4; ++nt) {
      oacc[0][nt] = wmma_bf16(Av0, Ba[nt], oacc[0][nt]);
      oacc[1][nt] = wmma_bf16(Av1, Ba[nt], oacc[1][nt]);
    }

    if (i0 + 32 < Nn) {  // prefetch next v rows -> global_prefetch_b8
      __builtin_prefetch(vb0 + i0 + 32, 0, 1);
      __builtin_prefetch(vb1 + i0 + 32, 0, 1);
    }
    Aq0 = AqN0; Aq1 = AqN1;
  }

  // ---- epilogue: out = gamma*acc + x ----
  const float gma = gamma_p[0];
  const float* xb = x + (size_t)b * Cc * Nn;
  float* ob = out + (size_t)b * Cc * Nn;
#pragma unroll
  for (int mg = 0; mg < 2; ++mg) {
#pragma unroll
    for (int nt = 0; nt < 4; ++nt) {
      const int j = j0 + 16 * nt + col;
#pragma unroll
      for (int r = 0; r < 8; ++r) {
        const int c = c0 + 128 * mg + 16 * w + 8 * g + r;
        const size_t idx = (size_t)c * Nn + j;
        ob[idx] = gma * oacc[mg][nt][r] + xb[idx];
      }
    }
  }
}

extern "C" void kernel_launch(void* const* d_in, const int* in_sizes, int n_in,
                              void* d_out, int out_size, void* d_ws, size_t ws_size,
                              hipStream_t stream) {
  const float* x     = (const float*)d_in[0];
  const float* wq    = (const float*)d_in[1];
  const float* bq    = (const float*)d_in[2];
  const float* wk    = (const float*)d_in[3];
  const float* bk    = (const float*)d_in[4];
  const float* wv    = (const float*)d_in[5];
  const float* bv    = (const float*)d_in[6];
  const float* gamma = (const float*)d_in[7];
  float* out = (float*)d_out;

  // Workspace carve-up (~76 MB total)
  char* p = (char*)d_ws;
  bf16*   xbT  = (bf16*)p;   p += (size_t)Bn * Nn * Cc * sizeof(bf16);   // 32 MB
  bf16*   qT   = (bf16*)p;   p += (size_t)Bn * Nn * CId * sizeof(bf16);  // 4 MB
  bf16*   kT   = (bf16*)p;   p += (size_t)Bn * Nn * CId * sizeof(bf16);  // 4 MB
  bf16*   vD   = (bf16*)p;   p += (size_t)Bn * Cc * Nn * sizeof(bf16);   // 32 MB
  bf16*   wqb  = (bf16*)p;   p += (size_t)CId * Cc * sizeof(bf16);
  bf16*   wkb  = (bf16*)p;   p += (size_t)CId * Cc * sizeof(bf16);
  bf16*   wvb  = (bf16*)p;   p += (size_t)Cc * Cc * sizeof(bf16);
  float2* stats = (float2*)p;

  cvt_x_transpose<<<dim3(Nn / 32, Cc / 32, Bn), dim3(32, 8), 0, stream>>>(x, xbT);
  cvt_w<<<(CId * Cc + 255) / 256, 256, 0, stream>>>(wq, wqb, CId * Cc);
  cvt_w<<<(CId * Cc + 255) / 256, 256, 0, stream>>>(wk, wkb, CId * Cc);
  cvt_w<<<(Cc * Cc + 255) / 256, 256, 0, stream>>>(wv, wvb, Cc * Cc);

  proj_qk<<<dim3(Nn / 64, Bn), 256, 0, stream>>>(xbT, wqb, bq, wkb, bk, qT, kT);
  proj_v<<<dim3(Nn / 64, Cc / 128, Bn), 256, 0, stream>>>(xbT, wvb, bv, vD);

  softmax_stats<<<dim3(Nn / 128, Bn), 256, 0, stream>>>(qT, kT, stats);
  attn_out<<<dim3(Nn / 64, Cc / 256, Bn), 256, 0, stream>>>(qT, kT, vD, stats, x,
                                                            gamma, out);
}
